// NaiveCustomLSTM_3822520893957
// MI455X (gfx1250) — matrix operand order repaired and checked by reading, not compiled, not run on hardware
//
#include <hip/hip_runtime.h>
#include <hip/hip_bf16.h>
#include <math.h>

// ---------------------------------------------------------------------------
// LSTM-style scan, CDNA5 (gfx1250, wave32, WMMA 16x16x32 bf16).
//   B=64, T=2048, I=H=256.  Gate f is dead code in the reference (c_new uses
//   h*c, not f*c) so we only compute gates i (U_i/V_i/b_i), g (U_c/V_c/b_c),
//   o (U_o/V_o/b_o).
//
// Phase 0: pack U,V into WMMA-B fragments [g][nT][kT][lane][16] bf16.
// Phase 1: WG = (t, 16-batch slab): stage x rows to LDS (bf16) once, 8 waves
//          compute 2 nT x 3 gates each with kT-double-buffered B fragments;
//          store accumulators in C/D-fragment layout P'[g][t][bT][nT][lane][8].
// Phase 2: persistent scan, 4 WGs x 16 batch rows. The wave's 48 V fragments
//          (384 VGPRs) are loaded ONCE into registers before the t-loop;
//          h/c state in registers; bf16 h mirror in LDS for the WMMA A operand.
//          Per step: 6 b128 P loads + 16 ds_load_b128 + 48 WMMAs.
// ---------------------------------------------------------------------------

typedef __attribute__((ext_vector_type(16))) __bf16 v16bf;
typedef __attribute__((ext_vector_type(8)))  __bf16 v8bf;
typedef __attribute__((ext_vector_type(8)))  float  v8f;

#define TT    2048
#define BB    64
#define HH    256
#define GATES 3

// Upk/Vpk: [3][16 nT][8 kT][32 lane][16] bf16 = 393216 B each
#define PACK_BYTES (GATES * 16 * 8 * 32 * 16 * 2)

// ---------------------------------------------------------------------------
// Phase 0: pack U/V into per-lane-contiguous WMMA B fragments.
// B-matrix (32x16, 16-bit): n = lane%16, k = kT*32 + (lane/16)*8 + j + (j>=8 ? 8 : 0), j=0..15
// ---------------------------------------------------------------------------
__global__ __launch_bounds__(256) void pack_kernel(
    const float* __restrict__ U0, const float* __restrict__ U1, const float* __restrict__ U2,
    const float* __restrict__ V0, const float* __restrict__ V1, const float* __restrict__ V2,
    __bf16* __restrict__ Upk, __bf16* __restrict__ Vpk)
{
    int idx = blockIdx.x * 256 + threadIdx.x;     // ((g*16+nT)*8+kT)*32+lane
    if (idx >= GATES * 16 * 8 * 32) return;
    int lane = idx & 31;
    int kT   = (idx >> 5) & 7;
    int nT   = (idx >> 8) & 15;
    int g    = idx >> 12;

    const float* U = (g == 0) ? U0 : ((g == 1) ? U1 : U2);
    const float* V = (g == 0) ? V0 : ((g == 1) ? V1 : V2);

    int n     = nT * 16 + (lane & 15);
    int kbase = kT * 32 + ((lane >> 4) << 3);

    __bf16* up = Upk + (size_t)idx * 16;
    __bf16* vp = Vpk + (size_t)idx * 16;
#pragma unroll
    for (int j = 0; j < 16; ++j) {
        up[j] = (__bf16)U[(size_t)(kbase + j + ((j >> 3) << 3)) * HH + n];
        vp[j] = (__bf16)V[(size_t)(kbase + j + ((j >> 3) << 3)) * HH + n];
    }
}

// ---------------------------------------------------------------------------
// Phase 1: P'[g][t][bT][nT][lane][8] = C/D fragments of bf16(x @ U_g + b_g).
// Grid: 8192 WGs = 2048 t x 4 bT. 8 waves x (2 nT x 3 gates) each.
// ---------------------------------------------------------------------------
__global__ __launch_bounds__(256) void proj_gemm_kernel(
    const float* __restrict__ x,          // [B, T, 256]
    const __bf16* __restrict__ Upk,
    const float* __restrict__ bi, const float* __restrict__ bg, const float* __restrict__ bo,
    __bf16* __restrict__ P)
{
    __shared__ __bf16 hx[16][HH];         // 16 rows of x at time t, bf16 (8 KB)

    const int t     = blockIdx.x >> 2;
    const int bT    = blockIdx.x & 3;
    const int bBase = bT * 16;
    const int tid   = threadIdx.x;

    // ---- stage x[bBase..+15, t, :] -> bf16 LDS (coalesced b128 loads)
    {
        const int i   = tid >> 4;         // row 0..15
        const int seg = tid & 15;         // 16-float segment
        const float4* src = (const float4*)(x + ((size_t)(bBase + i) * TT + t) * HH + seg * 16);
        float4 f0 = src[0], f1 = src[1], f2 = src[2], f3 = src[3];
        v8bf lo, hi;
        lo[0]=(__bf16)f0.x; lo[1]=(__bf16)f0.y; lo[2]=(__bf16)f0.z; lo[3]=(__bf16)f0.w;
        lo[4]=(__bf16)f1.x; lo[5]=(__bf16)f1.y; lo[6]=(__bf16)f1.z; lo[7]=(__bf16)f1.w;
        hi[0]=(__bf16)f2.x; hi[1]=(__bf16)f2.y; hi[2]=(__bf16)f2.z; hi[3]=(__bf16)f2.w;
        hi[4]=(__bf16)f3.x; hi[5]=(__bf16)f3.y; hi[6]=(__bf16)f3.z; hi[7]=(__bf16)f3.w;
        *(v8bf*)&hx[i][seg * 16]     = lo;
        *(v8bf*)&hx[i][seg * 16 + 8] = hi;
    }
    __syncthreads();

    const int lane   = tid & 31;
    const int wave   = tid >> 5;
    const int mlocal = lane & 15;
    const int koff   = (lane >> 4) << 3;  // A layout
    const size_t gstride = (size_t)16 * 8 * 32 * 16;

    v8f acc[2][GATES];
#pragma unroll
    for (int nt = 0; nt < 2; ++nt)
#pragma unroll
        for (int g = 0; g < GATES; ++g) acc[nt][g] = (v8f){};

    // kT-double-buffered B fragments (ping-pong on kT&1)
    v16bf bbuf[2][2][GATES];
#pragma unroll
    for (int nt = 0; nt < 2; ++nt) {
        const int nT = wave * 2 + nt;
        const size_t bidx = (((size_t)nT * 8 + 0) * 32 + lane) * 16;
#pragma unroll
        for (int g = 0; g < GATES; ++g)
            bbuf[0][nt][g] = *(const v16bf*)(Upk + (size_t)g * gstride + bidx);
    }

#pragma unroll
    for (int kT = 0; kT < 8; ++kT) {
        const int cur = kT & 1, nxt = cur ^ 1;
        if (kT < 7) {
#pragma unroll
            for (int nt = 0; nt < 2; ++nt) {
                const int nT = wave * 2 + nt;
                const size_t bidx = (((size_t)nT * 8 + (kT + 1)) * 32 + lane) * 16;
#pragma unroll
                for (int g = 0; g < GATES; ++g)
                    bbuf[nxt][nt][g] = *(const v16bf*)(Upk + (size_t)g * gstride + bidx);
            }
        }
        const __bf16* hp = &hx[mlocal][kT * 32 + koff];
        v8bf lo = *(const v8bf*)(hp);
        v8bf hi = *(const v8bf*)(hp + 16);
        v16bf a;
#pragma unroll
        for (int j = 0; j < 8; ++j) { a[j] = lo[j]; a[8 + j] = hi[j]; }

#pragma unroll
        for (int nt = 0; nt < 2; ++nt)
#pragma unroll
            for (int g = 0; g < GATES; ++g)
                acc[nt][g] = __builtin_amdgcn_wmma_f32_16x16x32_bf16(
                    false, a, false, bbuf[cur][nt][g], (short)0, acc[nt][g], false, false);
    }

    // ---- epilogue: bias + store fragments (16B contiguous per lane)
    const int N = lane & 15;
#pragma unroll
    for (int nt = 0; nt < 2; ++nt) {
        const int nT = wave * 2 + nt;
        const int n  = nT * 16 + N;
        const float b0 = bi[n], b1 = bg[n], b2 = bo[n];
#pragma unroll
        for (int g = 0; g < GATES; ++g) {
            const float bias = (g == 0) ? b0 : ((g == 1) ? b1 : b2);
            v8bf p;
#pragma unroll
            for (int v = 0; v < 8; ++v) p[v] = (__bf16)(acc[nt][g][v] + bias);
            size_t o = ((((size_t)g * TT + t) * 4 + bT) * 16 + nT) * 256 + lane * 8;
            *(v8bf*)(P + o) = p;
        }
    }
}

// ---------------------------------------------------------------------------
// Phase 2: persistent scan. 4 blocks x 256 threads (8 waves); block owns 16
// batch rows, wave owns nTiles {2w, 2w+1}. The wave's 48 V fragments are
// register-resident across the whole t-loop; h/c state in registers; bf16 h
// mirror in LDS for the WMMA A operand.
// ---------------------------------------------------------------------------
__global__ __launch_bounds__(256) void scan_kernel(
    const __bf16* __restrict__ P,
    const __bf16* __restrict__ Vpk,
    float* __restrict__ out)              // [B, T, H]
{
    __shared__ __bf16 hbf[16][HH];        // bf16 mirror of h (8 KB)

    const int lane  = threadIdx.x & 31;
    const int wave  = threadIdx.x >> 5;
    const int bT    = blockIdx.x;
    const int bBase = bT * 16;

    for (int i = threadIdx.x; i < 16 * HH; i += 256)
        ((__bf16*)hbf)[i] = (__bf16)0.0f;
    __syncthreads();

    const int mlocal = lane & 15;
    const int koff   = (lane >> 4) << 3;  // A layout
    const int N      = lane & 15;
    const int mbase  = (lane >> 4) << 3;  // C/D layout: M = v + 8*(lane/16)
    const size_t gstride = (size_t)16 * 8 * 32 * 16;

    // ---- load the wave's V working set ONCE: [nt][g][kT] -> 384 VGPRs
    v16bf vfrag[2][GATES][8];
#pragma unroll
    for (int nt = 0; nt < 2; ++nt) {
        const int nT = wave * 2 + nt;
#pragma unroll
        for (int g = 0; g < GATES; ++g)
#pragma unroll
            for (int kT = 0; kT < 8; ++kT) {
                const size_t bidx = (((size_t)nT * 8 + kT) * 32 + lane) * 16;
                vfrag[nt][g][kT] = *(const v16bf*)(Vpk + (size_t)g * gstride + bidx);
            }
    }

    float hreg[2][8];                     // owned h elements (f32)
    float creg[2][8];                     // owned c elements
#pragma unroll
    for (int nt = 0; nt < 2; ++nt)
#pragma unroll
        for (int v = 0; v < 8; ++v) { hreg[nt][v] = 0.0f; creg[nt][v] = 0.0f; }

    for (int t = 0; t < TT; ++t) {
        v8f acc[2][GATES];

        // ---- init accumulators from P' fragments: one b128 load each
#pragma unroll
        for (int nt = 0; nt < 2; ++nt) {
            const int nT = wave * 2 + nt;
#pragma unroll
            for (int g = 0; g < GATES; ++g) {
                size_t o = ((((size_t)g * TT + t) * 4 + bT) * 16 + nT) * 256 + lane * 8;
                v8bf frag = *(const v8bf*)(P + o);
#pragma unroll
                for (int v = 0; v < 8; ++v) acc[nt][g][v] = (float)frag[v];
            }
        }
        // prefetch next timestep's fragments
        if (t + 1 < TT) {
            size_t o = (((size_t)(t + 1) * 4 + bT) * 16 + wave * 2) * 256 + lane * 8;
            __builtin_prefetch(P + o, 0, 0);
        }

        // ---- h @ V_g : 8 k-steps, V operands already in registers
#pragma unroll
        for (int kT = 0; kT < 8; ++kT) {
            const __bf16* hp = &hbf[mlocal][kT * 32 + koff];
            v8bf lo = *(const v8bf*)(hp);
            v8bf hi = *(const v8bf*)(hp + 16);
            v16bf a;
#pragma unroll
            for (int j = 0; j < 8; ++j) { a[j] = lo[j]; a[8 + j] = hi[j]; }

#pragma unroll
            for (int nt = 0; nt < 2; ++nt)
#pragma unroll
                for (int g = 0; g < GATES; ++g)
                    acc[nt][g] = __builtin_amdgcn_wmma_f32_16x16x32_bf16(
                        false, a, false, vfrag[nt][g][kT], (short)0, acc[nt][g], false, false);
        }
        __syncthreads();   // all WMMA reads of hbf done before updates

        // ---- elementwise gate math; h/c state stays in registers
#pragma unroll
        for (int nt = 0; nt < 2; ++nt) {
            const int n = (wave * 2 + nt) * 16 + N;
#pragma unroll
            for (int v = 0; v < 8; ++v) {
                const int M = mbase + v;
                float it = 1.0f / (1.0f + __expf(-acc[nt][0][v]));
                float gt = tanhf(acc[nt][1][v]);
                float ot = 1.0f / (1.0f + __expf(-acc[nt][2][v]));
                float cnew = hreg[nt][v] * creg[nt][v] + it * gt;  // h*c + i*g
                float hnew = ot * tanhf(cnew);
                creg[nt][v] = cnew;
                hreg[nt][v] = hnew;
                hbf[M][n]   = (__bf16)hnew;
                out[((size_t)(bBase + M) * TT + t) * HH + n] = hnew;
            }
        }
        __syncthreads();   // updates visible before next step's WMMA reads
    }
}

// ---------------------------------------------------------------------------
extern "C" void kernel_launch(void* const* d_in, const int* in_sizes, int n_in,
                              void* d_out, int out_size, void* d_ws, size_t ws_size,
                              hipStream_t stream)
{
    const float* x   = (const float*)d_in[0];
    const float* U_i = (const float*)d_in[1];
    const float* V_i = (const float*)d_in[2];
    const float* b_i = (const float*)d_in[3];
    // d_in[4..6] = U_f, V_f, b_f : dead code in the reference (f_t unused)
    const float* U_c = (const float*)d_in[7];
    const float* V_c = (const float*)d_in[8];
    const float* b_c = (const float*)d_in[9];
    const float* U_o = (const float*)d_in[10];
    const float* V_o = (const float*)d_in[11];
    const float* b_o = (const float*)d_in[12];

    char* ws = (char*)d_ws;
    __bf16* Upk = (__bf16*)(ws);
    __bf16* Vpk = (__bf16*)(ws + (size_t)PACK_BYTES);
    __bf16* P   = (__bf16*)(ws + 2 * (size_t)PACK_BYTES);   // ~201 MB

    pack_kernel<<<(GATES * 16 * 8 * 32 + 255) / 256, 256, 0, stream>>>(
        U_i, U_c, U_o, V_i, V_c, V_o, Upk, Vpk);

    proj_gemm_kernel<<<TT * 4, 256, 0, stream>>>(x, Upk, b_i, b_c, b_o, P);

    scan_kernel<<<4, 256, 0, stream>>>(P, Vpk, (float*)d_out);
}
